// SelfAttention_35046933135796
// MI455X (gfx1250) — matrix operand order repaired and checked by reading, not compiled, not run on hardware
//
#include <hip/hip_runtime.h>
#include <stdint.h>

// ---------------------------------------------------------------------------
// Problem constants (B=4, S=2048, E=1024, H=16, HD=64)
// ---------------------------------------------------------------------------
#define BB 4
#define SS 2048
#define EE 1024
#define HH 16
#define HDD 64
#define MROWS (BB * SS)      // 8192 tokens
#define N_QKV (3 * EE)       // 3072

typedef __attribute__((ext_vector_type(16))) __bf16 v16bf;
typedef __attribute__((ext_vector_type(4)))  __bf16 v4bf;
typedef __attribute__((ext_vector_type(8)))  float  v8f;
typedef __attribute__((ext_vector_type(4)))  unsigned int u32x4;
typedef __attribute__((ext_vector_type(8)))  int i32x8;
typedef __attribute__((ext_vector_type(4)))  int i32x4;

#define DEV __device__ __forceinline__

// Tensor Data Mover builtin exists on both probed toolchains but with different
// arity: 5 args on ROCm 7.2 (clang-22), 6 args on amdgpu-toolchain (clang-23+).
#if defined(__has_builtin)
#if __has_builtin(__builtin_amdgcn_tensor_load_to_lds)
#define USE_TDM 1
#endif
#endif
#ifndef USE_TDM
#define USE_TDM 0
#endif

union FragU { v16bf v; uint4 q[2]; };

DEV v16bf load_frag2(const __bf16* p0, const __bf16* p1) {
  FragU f;
  f.q[0] = *reinterpret_cast<const uint4*>(p0);
  f.q[1] = *reinterpret_cast<const uint4*>(p1);
  return f.v;
}

// A-matrix fragment (16x32 bf16, MxK): lane half h holds K = {k0+h*8..+7, k0+16+h*8..+7}
DEV v16bf load_afrag(const __bf16* row, int k0, int hl) {
  return load_frag2(row + k0 + hl * 8, row + k0 + 16 + hl * 8);
}
// B-matrix fragment (32x16 bf16, KxN): lane half h holds K = {k0+h*8..+7, k0+16+h*8..+7}
DEV v16bf load_bfrag(const __bf16* row, int k0, int hl) {
  return load_frag2(row + k0 + hl * 8, row + k0 + 16 + hl * 8);
}

DEV v8f wmma_bf16(v16bf a, v16bf b, v8f c) {
  return __builtin_amdgcn_wmma_f32_16x16x32_bf16(false, a, false, b, (short)0, c,
                                                 false, false);
}

DEV float redmax16(float x) {
#pragma unroll
  for (int m = 1; m < 16; m <<= 1) x = fmaxf(x, __shfl_xor(x, m, 32));
  return x;
}
DEV float redsum16(float x) {
#pragma unroll
  for (int m = 1; m < 16; m <<= 1) x += __shfl_xor(x, m, 32);
  return x;
}

// ---------------------------------------------------------------------------
// TDM 2-D tile load descriptor (ISA cdna5 ch.8): bf16 elements (data_size=1->2B),
// tile = tile_w x tile_h, tensor row stride = stride_elems.
// ---------------------------------------------------------------------------
#if USE_TDM
DEV void tdm_load_2d(unsigned lds_addr, const void* gptr, unsigned tile_w,
                     unsigned tile_h, unsigned stride_elems, unsigned tensor_w,
                     unsigned tensor_h) {
  unsigned long long ga = (unsigned long long)(uintptr_t)gptr;
  u32x4 g0;
  g0[0] = 1u;                                                 // count=1, user mode
  g0[1] = lds_addr;                                           // LDS byte address
  g0[2] = (unsigned)ga;                                       // global_addr[31:0]
  g0[3] = (unsigned)((ga >> 32) & 0x01FFFFFFu) | (2u << 30);  // addr[56:32] | type=2
  i32x8 g1;
  g1[0] = (int)(1u << 16);                                  // data_size=1 (2 bytes)
  g1[1] = (int)((tensor_w & 0xFFFFu) << 16);                // tensor_dim0[15:0]
  g1[2] = (int)((tensor_w >> 16) | ((tensor_h & 0xFFFFu) << 16));
  g1[3] = (int)((tensor_h >> 16) | (tile_w << 16));         // tile_dim0
  g1[4] = (int)tile_h;                                      // tile_dim1, tile_dim2=0
  g1[5] = (int)stride_elems;                                // tensor_dim0_stride lo
  g1[6] = 0;
  g1[7] = 0;
  i32x4 zz = {0, 0, 0, 0};
#if defined(__clang_major__) && (__clang_major__ >= 23)
  i32x8 zz8 = {0, 0, 0, 0, 0, 0, 0, 0};
  __builtin_amdgcn_tensor_load_to_lds(g0, g1, zz, zz, zz8, 0);
#else
  __builtin_amdgcn_tensor_load_to_lds(g0, g1, zz, zz, 0);
#endif
}
DEV unsigned lds_addr_of(const void* p) { return (unsigned)(uintptr_t)p; }
#endif

// ---------------------------------------------------------------------------
// Kernel 0: f32 -> bf16 conversion, 4 elements per thread
// ---------------------------------------------------------------------------
__global__ void cvt_f32_bf16(const float* __restrict__ src, __bf16* __restrict__ dst,
                             int n) {
  int i = (blockIdx.x * blockDim.x + threadIdx.x) * 4;
  if (i + 3 < n) {
    float4 f = *reinterpret_cast<const float4*>(src + i);
    v4bf o;
    o.x = (__bf16)f.x; o.y = (__bf16)f.y; o.z = (__bf16)f.z; o.w = (__bf16)f.w;
    *reinterpret_cast<v4bf*>(dst + i) = o;
  }
}

// ---------------------------------------------------------------------------
// Shared GEMM core: C_block[128x64] += A[128x1024] @ B[64x1024]^T for one block.
// LDS double-buffered 64-wide K stages; TDM (wave 0) or batched cooperative copy.
// 8 waves: wave tile = 64x16 (wm = wave>>2, wn = wave&3), 8 WMMAs/stage/wave.
// Fragments loaded up-front so dscnt waits can retire incrementally.
// ---------------------------------------------------------------------------
DEV void gemm_stage_compute(const __bf16* curA, const __bf16* curB, int wm, int wn,
                            int hl, int r, v8f acc[4]) {
  const __bf16* bL = curB + (wn * 16 + r) * 64;
  v16bf bf0 = load_bfrag(bL, 0, hl);
  v16bf bf1 = load_bfrag(bL, 32, hl);
  v16bf af0[4], af1[4];
#pragma unroll
  for (int t = 0; t < 4; ++t) {
    const __bf16* aL = curA + (wm * 64 + t * 16 + r) * 64;
    af0[t] = load_afrag(aL, 0, hl);
    af1[t] = load_afrag(aL, 32, hl);
  }
#pragma unroll
  for (int t = 0; t < 4; ++t) acc[t] = wmma_bf16(af0[t], bf0, acc[t]);
#pragma unroll
  for (int t = 0; t < 4; ++t) acc[t] = wmma_bf16(af1[t], bf1, acc[t]);
}

DEV void gemm_tile(const __bf16* __restrict__ Ag, const __bf16* __restrict__ Bg,
                   __bf16* ldsA, __bf16* ldsB, int m0, int n0b,
                   unsigned tensorA_h, unsigned tensorB_h, v8f acc[4]) {
  const int tid = threadIdx.x;
  const int lane = tid & 31;
  const int wave = tid >> 5;
  const int hl = lane >> 4;
  const int r = lane & 15;
  const int wm = wave >> 2;
  const int wn = wave & 3;
  const int NSTAGE = EE / 64;  // 16

#if USE_TDM
  if (wave == 0) {  // one wave issues DMA; TDM ignores EXEC, tracked by TENSORcnt
    tdm_load_2d(lds_addr_of(ldsA), Ag + (size_t)m0 * EE, 64, 128, EE, EE, tensorA_h);
    tdm_load_2d(lds_addr_of(ldsB), Bg + (size_t)n0b * EE, 64, 64, EE, EE, tensorB_h);
  }
  for (int s = 0; s < NSTAGE; ++s) {
    __bf16* curA = ldsA + (s & 1) * (128 * 64);
    __bf16* curB = ldsB + (s & 1) * (64 * 64);
    if (wave == 0) {
      if (s + 1 < NSTAGE) {
        __bf16* nxtA = ldsA + ((s + 1) & 1) * (128 * 64);
        __bf16* nxtB = ldsB + ((s + 1) & 1) * (64 * 64);
        tdm_load_2d(lds_addr_of(nxtA), Ag + (size_t)m0 * EE + (s + 1) * 64, 64, 128,
                    EE, EE, tensorA_h);
        tdm_load_2d(lds_addr_of(nxtB), Bg + (size_t)n0b * EE + (s + 1) * 64, 64, 64,
                    EE, EE, tensorB_h);
        __builtin_amdgcn_s_wait_tensorcnt(2);  // in-order: stage s has landed
      } else {
        __builtin_amdgcn_s_wait_tensorcnt(0);
      }
    }
    __syncthreads();
    gemm_stage_compute(curA, curB, wm, wn, hl, r, acc);
    __syncthreads();
  }
#else
  for (int s = 0; s < NSTAGE; ++s) {
    __bf16* curA = ldsA + (s & 1) * (128 * 64);
    __bf16* curB = ldsB + (s & 1) * (64 * 64);
    // Batched staging: issue all global loads, then all LDS stores.
    uint4 ta[4], tb[2];
#pragma unroll
    for (int ii = 0; ii < 4; ++ii) {  // A: 128 rows x 64 cols = 1024 16B chunks
      int i = tid + ii * 256;
      int row = i >> 3, c = i & 7;
      ta[ii] = *reinterpret_cast<const uint4*>(Ag + (size_t)(m0 + row) * EE + s * 64 +
                                               c * 8);
    }
#pragma unroll
    for (int ii = 0; ii < 2; ++ii) {  // B: 64 rows x 64 cols = 512 16B chunks
      int i = tid + ii * 256;
      int row = i >> 3, c = i & 7;
      tb[ii] = *reinterpret_cast<const uint4*>(Bg + (size_t)(n0b + row) * EE + s * 64 +
                                               c * 8);
    }
#pragma unroll
    for (int ii = 0; ii < 4; ++ii) {
      int i = tid + ii * 256;
      int row = i >> 3, c = i & 7;
      *reinterpret_cast<uint4*>(curA + row * 64 + c * 8) = ta[ii];
    }
#pragma unroll
    for (int ii = 0; ii < 2; ++ii) {
      int i = tid + ii * 256;
      int row = i >> 3, c = i & 7;
      *reinterpret_cast<uint4*>(curB + row * 64 + c * 8) = tb[ii];
    }
    __syncthreads();
    gemm_stage_compute(curA, curB, wm, wn, hl, r, acc);
    __syncthreads();
  }
#endif
}

// ---------------------------------------------------------------------------
// Kernel 1: QKV projection with Q/K/Vt scatter epilogue
// ---------------------------------------------------------------------------
__global__ __launch_bounds__(256) void qkv_gemm_kernel(
    const __bf16* __restrict__ X, const __bf16* __restrict__ W,
    __bf16* __restrict__ Qb, __bf16* __restrict__ Kb, __bf16* __restrict__ Vt) {
  __shared__ __align__(16) __bf16 ldsA[2 * 128 * 64];
  __shared__ __align__(16) __bf16 ldsB[2 * 64 * 64];

  const int lane = threadIdx.x & 31;
  const int wave = threadIdx.x >> 5;
  const int hl = lane >> 4;
  const int r = lane & 15;
  const int m0blk = blockIdx.y * 128;
  const int n0blk = blockIdx.x * 64;

  const v8f z8 = {0.f, 0.f, 0.f, 0.f, 0.f, 0.f, 0.f, 0.f};
  v8f acc[4] = {z8, z8, z8, z8};

  gemm_tile(X, W, ldsA, ldsB, m0blk, n0blk, MROWS, N_QKV, acc);

  // Scatter: n -> (head h, within-head rr); rr: 0-63 Q, 64-127 K, 128-191 V(transposed)
  const int m0 = m0blk + (wave >> 2) * 64;
  const int n = n0blk + (wave & 3) * 16 + r;
  const int h = n / 192;
  const int rr = n - h * 192;
#pragma unroll
  for (int t = 0; t < 4; ++t) {
#pragma unroll
    for (int j = 0; j < 8; ++j) {
      const int tok = m0 + t * 16 + hl * 8 + j;
      const int b = tok >> 11;
      const int s = tok & (SS - 1);
      const int bh = b * HH + h;
      const __bf16 val = (__bf16)acc[t][j];
      if (rr < 64)
        Qb[((size_t)bh * SS + s) * HDD + rr] = val;
      else if (rr < 128)
        Kb[((size_t)bh * SS + s) * HDD + (rr - 64)] = val;
      else
        Vt[((size_t)bh * HDD + (rr - 128)) * SS + s] = val;
    }
  }
}

// ---------------------------------------------------------------------------
// Kernel 2: causal flash attention.  grid = (S/128, B*H), 8 waves/block.
// Per-wave 16 query rows, 32-key blocks, online softmax, LDS P-rotation.
// ---------------------------------------------------------------------------
__global__ __launch_bounds__(256) void flash_attn_kernel(
    const __bf16* __restrict__ Qb, const __bf16* __restrict__ Kb,
    const __bf16* __restrict__ Vt, __bf16* __restrict__ O) {
  __shared__ __align__(16) __bf16 ldsP[8][16 * 32];

  const int lane = threadIdx.x & 31;
  const int wave = threadIdx.x >> 5;
  const int hl = lane >> 4;
  const int r = lane & 15;
  const int bh = blockIdx.y;
  const int qbase = blockIdx.x * 128 + wave * 16;

  const __bf16* Qh = Qb + (size_t)bh * (SS * HDD);
  const __bf16* Kh = Kb + (size_t)bh * (SS * HDD);
  const __bf16* Vh = Vt + (size_t)bh * (HDD * SS);

  const __bf16* qrow = Qh + (size_t)(qbase + r) * HDD;
  v16bf qf0 = load_afrag(qrow, 0, hl);
  v16bf qf1 = load_afrag(qrow, 32, hl);

  const v8f z8 = {0.f, 0.f, 0.f, 0.f, 0.f, 0.f, 0.f, 0.f};
  v8f o0 = z8, o1 = z8, o2 = z8, o3 = z8;
  float m_run[8], l_run[8];
#pragma unroll
  for (int j = 0; j < 8; ++j) { m_run[j] = -1e30f; l_run[j] = 0.f; }

  __bf16* pbuf = &ldsP[wave][0];
  const int nkb = (qbase + 15) / 32 + 1;

  for (int kb = 0; kb < nkb; ++kb) {
    const int kbase = kb * 32;
    const __bf16* krow0 = Kh + (size_t)(kbase + r) * HDD;
    const __bf16* krow1 = Kh + (size_t)(kbase + 16 + r) * HDD;

    v8f s0 = z8, s1 = z8;
    {
      v16bf b00 = load_bfrag(krow0, 0, hl);
      v16bf b01 = load_bfrag(krow1, 0, hl);
      s0 = wmma_bf16(qf0, b00, s0);
      s1 = wmma_bf16(qf0, b01, s1);
      v16bf b10 = load_bfrag(krow0, 32, hl);
      v16bf b11 = load_bfrag(krow1, 32, hl);
      s0 = wmma_bf16(qf1, b10, s0);
      s1 = wmma_bf16(qf1, b11, s1);
    }

    // V fragments don't depend on scores: issue loads before softmax to overlap
    v16bf vf0 = load_bfrag(Vh + (size_t)(0 * 16 + r) * SS, kbase, hl);
    v16bf vf1 = load_bfrag(Vh + (size_t)(1 * 16 + r) * SS, kbase, hl);
    v16bf vf2 = load_bfrag(Vh + (size_t)(2 * 16 + r) * SS, kbase, hl);
    v16bf vf3 = load_bfrag(Vh + (size_t)(3 * 16 + r) * SS, kbase, hl);
    if (kb + 1 < nkb) {
      __builtin_prefetch(Kh + (size_t)(kbase + 32 + r) * HDD, 0, 3);
      __builtin_prefetch(Vh + (size_t)r * SS + kbase + 32, 0, 3);
    }

    float p0[8], p1[8];
#pragma unroll
    for (int j = 0; j < 8; ++j) {
      const int q = qbase + hl * 8 + j;
      float a = s0[j] * 0.125f;
      float b = s1[j] * 0.125f;
      if (kbase + r > q) a = -1e30f;
      if (kbase + 16 + r > q) b = -1e30f;
      float mx = redmax16(fmaxf(a, b));
      float mnew = fmaxf(m_run[j], mx);
      float alpha = __expf(m_run[j] - mnew);
      float e0 = __expf(a - mnew);
      float e1 = __expf(b - mnew);
      float rs = redsum16(e0 + e1);
      l_run[j] = l_run[j] * alpha + rs;
      m_run[j] = mnew;
      o0[j] *= alpha; o1[j] *= alpha; o2[j] *= alpha; o3[j] *= alpha;
      p0[j] = e0; p1[j] = e1;
    }

    // rotate P (16x32) C-layout -> A-layout through per-wave private LDS
#pragma unroll
    for (int j = 0; j < 8; ++j) {
      pbuf[(hl * 8 + j) * 32 + r] = (__bf16)p0[j];
      pbuf[(hl * 8 + j) * 32 + 16 + r] = (__bf16)p1[j];
    }
    v16bf pf = load_afrag(pbuf + r * 32, 0, hl);

    o0 = wmma_bf16(pf, vf0, o0);
    o1 = wmma_bf16(pf, vf1, o1);
    o2 = wmma_bf16(pf, vf2, o2);
    o3 = wmma_bf16(pf, vf3, o3);
  }

  const int b = bh >> 4;
  const int h = bh & 15;
  __bf16* orow0 = O + (size_t)(b * SS + qbase) * EE + h * HDD;
#pragma unroll
  for (int j = 0; j < 8; ++j) {
    const float inv = 1.f / l_run[j];
    __bf16* orow = orow0 + (size_t)(hl * 8 + j) * EE;
    orow[0 + r]  = (__bf16)(o0[j] * inv);
    orow[16 + r] = (__bf16)(o1[j] * inv);
    orow[32 + r] = (__bf16)(o2[j] * inv);
    orow[48 + r] = (__bf16)(o3[j] * inv);
  }
}

// ---------------------------------------------------------------------------
// Kernel 3: output projection + bias, f32 to d_out
// ---------------------------------------------------------------------------
__global__ __launch_bounds__(256) void out_proj_kernel(
    const __bf16* __restrict__ A, const __bf16* __restrict__ W,
    const float* __restrict__ bias, float* __restrict__ C) {
  __shared__ __align__(16) __bf16 ldsA[2 * 128 * 64];
  __shared__ __align__(16) __bf16 ldsB[2 * 64 * 64];

  const int lane = threadIdx.x & 31;
  const int wave = threadIdx.x >> 5;
  const int hl = lane >> 4;
  const int r = lane & 15;
  const int m0blk = blockIdx.y * 128;
  const int n0blk = blockIdx.x * 64;

  const v8f z8 = {0.f, 0.f, 0.f, 0.f, 0.f, 0.f, 0.f, 0.f};
  v8f acc[4] = {z8, z8, z8, z8};

  gemm_tile(A, W, ldsA, ldsB, m0blk, n0blk, MROWS, EE, acc);

  const int m0 = m0blk + (wave >> 2) * 64;
  const int n = n0blk + (wave & 3) * 16 + r;
  const float bn = bias[n];
#pragma unroll
  for (int t = 0; t < 4; ++t) {
#pragma unroll
    for (int j = 0; j < 8; ++j) {
      const int tok = m0 + t * 16 + hl * 8 + j;
      C[(size_t)tok * EE + n] = acc[t][j] + bn;
    }
  }
}

// ---------------------------------------------------------------------------
// Launch
// ---------------------------------------------------------------------------
extern "C" void kernel_launch(void* const* d_in, const int* in_sizes, int n_in,
                              void* d_out, int out_size, void* d_ws, size_t ws_size,
                              hipStream_t stream) {
  (void)in_sizes; (void)n_in; (void)out_size; (void)ws_size;
  const float* x    = (const float*)d_in[0];   // [B,S,E]
  const float* Wqkv = (const float*)d_in[1];   // [3E,E]
  const float* Wo   = (const float*)d_in[2];   // [E,E]
  const float* bo   = (const float*)d_in[3];   // [E]
  float* out = (float*)d_out;                  // [B,S,E]

  char* ws = (char*)d_ws;
  size_t off = 0;
  auto take = [&](size_t bytes) -> void* {
    void* p = ws + off;
    off += (bytes + 255) & ~(size_t)255;
    return p;
  };
  __bf16* xb    = (__bf16*)take((size_t)MROWS * EE * 2);
  __bf16* wqkvb = (__bf16*)take((size_t)N_QKV * EE * 2);
  __bf16* wob   = (__bf16*)take((size_t)EE * EE * 2);
  __bf16* Qb    = (__bf16*)take((size_t)BB * HH * SS * HDD * 2);
  __bf16* Kb    = (__bf16*)take((size_t)BB * HH * SS * HDD * 2);
  __bf16* Vt    = (__bf16*)take((size_t)BB * HH * HDD * SS * 2);
  __bf16* Ob    = (__bf16*)take((size_t)MROWS * EE * 2);

  cvt_f32_bf16<<<dim3(MROWS * EE / 4 / 256), dim3(256), 0, stream>>>(x, xb, MROWS * EE);
  cvt_f32_bf16<<<dim3(N_QKV * EE / 4 / 256), dim3(256), 0, stream>>>(Wqkv, wqkvb,
                                                                     N_QKV * EE);
  cvt_f32_bf16<<<dim3(EE * EE / 4 / 256), dim3(256), 0, stream>>>(Wo, wob, EE * EE);

  qkv_gemm_kernel<<<dim3(N_QKV / 64, MROWS / 128), dim3(256), 0, stream>>>(
      xb, wqkvb, Qb, Kb, Vt);

  flash_attn_kernel<<<dim3(SS / 128, BB * HH), dim3(256), 0, stream>>>(Qb, Kb, Vt, Ob);

  out_proj_kernel<<<dim3(EE / 64, MROWS / 128), dim3(256), 0, stream>>>(Ob, wob, bo,
                                                                        out);
}
